// RNN_LSTM_22703197127160
// MI455X (gfx1250) — matrix operand order repaired and checked by reading, not compiled, not run on hardware
//
#include <hip/hip_runtime.h>

// ---------------------------------------------------------------------------
// 3-layer LSTM + dense head for MI455X (gfx1250), wave32 WMMA f16->f32.
//
//  - Input projections X@W / final H@Wd: large parallel WMMA GEMMs.
//  - Recurrence: ONE persistent kernel per layer (32 co-resident blocks,
//    device-scope atomic step barrier).  U tiles staged in LDS once with a
//    bank-conflict-optimal 48B lane stride; h_{t-1} staged in LDS per step
//    via gfx1250 async global->LDS (overlapped with the Zpre loads); the
//    WMMA loop double-buffers ds_load_b128 fragments so LDS latency hides
//    behind v_wmma instead of serializing with it.
// ---------------------------------------------------------------------------

typedef __attribute__((ext_vector_type(16))) _Float16 v16h;
typedef __attribute__((ext_vector_type(8)))  float    v8f;
typedef int v4i_vs __attribute__((vector_size(16)));   // matches async-LDS builtin param

#define Bb 32
#define Tt 512
#define Dd 256
#define Uu 512
#define G4 2048             // 4*U
#define MROWS (Bb * Tt)     // 16384
#define NBLK 32             // persistent blocks per recurrence layer

#define HROW 520            // padded LDS row stride for h (f16): 260 dwords -> start bank 4*r
#define BLSTR 24            // padded LDS lane stride for B tiles (f16): 48B -> start bank 12*l

#if __has_builtin(__builtin_amdgcn_global_load_async_to_lds_b128) && \
    __has_builtin(__builtin_amdgcn_s_wait_asynccnt)
#define ASYNC_LDS 1
#endif

static __device__ __forceinline__ int lane_id() { return threadIdx.x & 31; }

// ---------------------------------------------------------------------------
// Pack W[K x N] (f32 row-major) into per-tile B-fragment layout, f16.
// Tile (kt, nt) stored at ((nt*KT + kt)*32 + lane)*16 f16:
//   lanes 0-15  hold K = kt*32 + {0..7, 16..23},  N = lane
//   lanes 16-31 hold K = kt*32 + {8..15, 24..31}, N = lane-16
// ---------------------------------------------------------------------------
__global__ void pack_b(const float* __restrict__ W, _Float16* __restrict__ out,
                       int K, int N) {
    int KT = K >> 5, NT = N >> 4;
    int wave = threadIdx.x >> 5;
    int tile = blockIdx.x * 8 + wave;
    if (tile >= KT * NT) return;
    int nt = tile / KT, kt = tile - nt * KT;
    int l  = lane_id();
    int n  = nt * 16 + (l & 15);
    int kb = kt * 32 + (l >> 4) * 8;
    _Float16* dst = out + ((size_t)tile * 32 + l) * 16;
#pragma unroll
    for (int j = 0; j < 8; ++j) {
        dst[2 * j]     = (_Float16)W[(size_t)(kb + 2 * j + (j & 4) * 2)     * N + n];
        dst[2 * j + 1] = (_Float16)W[(size_t)(kb + 2 * j + (j & 4) * 2 + 1) * N + n];
    }
}

// ---------------------------------------------------------------------------
// Generic GEMM: C[M x N] = A[M x K] @ Bpacked + bias.  One 16x16 tile/wave,
// 8 waves/block.  A is row-major f32 or f16; C is f16 or f32.
// A-fragment (16-bit A 16x32): lane l, r=l&15 (row), hi=l>>4;
//   vgprs 0-3: K = hi*8 + {0..7};  vgprs 4-7: K = 16 + hi*8 + {0..7}
// ---------------------------------------------------------------------------
template <bool A_IS_F32, bool OUT_F16>
__global__ void gemm_wmma(const void* __restrict__ Aptr,
                          const _Float16* __restrict__ Bp,
                          const float* __restrict__ bias,
                          void* __restrict__ Cptr,
                          int M, int N, int K, int lda) {
    int KT = K >> 5, NTt = N >> 4;
    int wave = threadIdx.x >> 5;
    int tile = blockIdx.x * 8 + wave;
    int mt = tile / NTt, nt = tile - mt * NTt;
    if (mt >= (M >> 4)) return;
    int l = lane_id();
    int r = l & 15, hi = l >> 4;
    int arow = mt * 16 + r;

    v8f acc = {};
    const float*    Af = (const float*)Aptr;
    const _Float16* Ah = (const _Float16*)Aptr;

    for (int kt = 0; kt < KT; ++kt) {
        v16h a, b;
        int kb = kt * 32 + hi * 8;
        if (A_IS_F32) {
            const float* ap = Af + (size_t)arow * lda + kb;
#pragma unroll
            for (int j = 0; j < 4; ++j) {
                float2 p0 = *(const float2*)(ap + 2 * j);
                float2 p1 = *(const float2*)(ap + 16 + 2 * j);
                a[2 * j]     = (_Float16)p0.x;  a[2 * j + 1]     = (_Float16)p0.y;
                a[8 + 2 * j] = (_Float16)p1.x;  a[8 + 2 * j + 1] = (_Float16)p1.y;
            }
        } else {
            const _Float16* ap = Ah + (size_t)arow * lda + kb;
            uint32_t* av = (uint32_t*)&a;
#pragma unroll
            for (int j = 0; j < 4; ++j) {
                av[j]     = *(const uint32_t*)(ap + 2 * j);
                av[4 + j] = *(const uint32_t*)(ap + 16 + 2 * j);
            }
        }
        const uint4* bsrc = (const uint4*)(Bp + ((size_t)(nt * KT + kt) * 32 + l) * 16);
        uint4* bv = (uint4*)&b;
        bv[0] = bsrc[0];
        bv[1] = bsrc[1];
        acc = __builtin_amdgcn_wmma_f32_16x16x32_f16(false, a, false, b,
                                                     (short)0, acc, false, false);
    }

    // C/D layout: vgpr v -> M = hi*8 + v ; N = l&15
#pragma unroll
    for (int v = 0; v < 8; ++v) {
        int row = mt * 16 + hi * 8 + v;
        int col = nt * 16 + r;
        float val = acc[v] + (bias ? bias[col] : 0.0f);
        if (OUT_F16) ((_Float16*)Cptr)[(size_t)row * N + col] = (_Float16)val;
        else         ((float*)Cptr)[(size_t)row * N + col] = val;
    }
}

// ---------------------------------------------------------------------------
// Init per-layer recurrent state from broadcast h0/c0 [1 x U]; reset barrier.
// ---------------------------------------------------------------------------
__global__ void init_state(const float* __restrict__ h0, const float* __restrict__ c0,
                           _Float16* __restrict__ h_state, float* __restrict__ c_state,
                           unsigned int* __restrict__ barrier_ctr) {
    int i = blockIdx.x * blockDim.x + threadIdx.x;   // 0 .. B*U-1
    if (i == 0) *barrier_ctr = 0u;
    int u = i & (Uu - 1);
    h_state[i] = (_Float16)h0[u];
    c_state[i] = c0[u];
}

// ---------------------------------------------------------------------------
// Persistent LSTM layer.  grid = NBLK = 32 blocks, 256 threads (8 waves).
// Block owns hidden units [blockIdx*16, +16) for ALL timesteps.
// ---------------------------------------------------------------------------
__global__ void lstm_layer(const _Float16* __restrict__ Z,     // [B*T, 4U] f16
                           const _Float16* __restrict__ Up,    // packed U [U x 4U]
                           const float* __restrict__ bias,     // [4U]
                           _Float16* __restrict__ hb0,         // [B, U] f16 (t even in)
                           _Float16* __restrict__ hb1,         // [B, U] f16 (t odd in)
                           float* __restrict__ c_state,        // [B, U] f32
                           _Float16* __restrict__ Hseq,        // [B*T, U] f16
                           unsigned int* __restrict__ barrier_ctr) {
    // 64 B tiles (4 gates x 16 ktiles), 32 lanes x 24 f16 (16 used, 48B stride)
    __shared__ _Float16 uB[64 * 32 * BLSTR];        // 96 KB
    __shared__ _Float16 hA[Bb * HROW];              // 33.3 KB (row stride 520)
    __shared__ float    zbuf[8 * 256];              // 8 KB

    const int KT = Uu >> 5;               // 16 k-tiles
    int wave = threadIdx.x >> 5;
    int l = lane_id();
    int r = l & 15, hi = l >> 4;
    int mt = wave & 1;                    // batch tile (0..1)
    int gt = wave >> 1;                   // gate (i,f,g,o)
    int ncol = gt * Uu + blockIdx.x * 16; // column in [0, 4U)
    int arow = mt * 16 + r;

    const float bval = bias[ncol + r];    // loop-invariant bias for this lane

    // ---- stage this block's 64 U tiles into LDS once (8 tiles per wave) ----
    for (int i = wave; i < 64; i += 8) {
        int g  = i >> 4, kt = i & 15;
        int nt = (g * Uu + blockIdx.x * 16) >> 4;
        const uint4* src = (const uint4*)(Up + ((size_t)(nt * KT + kt) * 32 + l) * 16);
        char* dst = (char*)uB + ((size_t)i * 32 + l) * (BLSTR * 2);
        uint4 d0 = src[0], d1 = src[1];
        *(uint4*)dst        = d0;
        *(uint4*)(dst + 16) = d1;
    }
    __syncthreads();

    // fragment loader: A from hA (padded rows), B from uB (48B lane stride)
    auto load_frag = [&](int kt, v16h& a, v16h& b) {
        const _Float16* ap = hA + arow * HROW + kt * 32 + hi * 8;
        uint32_t* av = (uint32_t*)&a;
#pragma unroll
        for (int j = 0; j < 4; ++j) {
            av[j]     = *(const uint32_t*)(ap + 2 * j);
            av[4 + j] = *(const uint32_t*)(ap + 16 + 2 * j);
        }
        const char* bp = (const char*)uB + ((size_t)(gt * 16 + kt) * 32 + l) * (BLSTR * 2);
        uint4* bv = (uint4*)&b;
        bv[0] = *(const uint4*)bp;
        bv[1] = *(const uint4*)(bp + 16);
    };

    for (int t = 0; t < Tt; ++t) {
        const _Float16* h_in  = (t & 1) ? hb1 : hb0;
        _Float16*       h_out = (t & 1) ? hb0 : hb1;

        // ---- issue h_{t-1} [32 x 512] f16 staging into LDS (padded rows) ----
        {
            int row = threadIdx.x >> 3;           // 0..31
            int cb  = (threadIdx.x & 7) * 64;     // 64 f16 per thread
            const char* gsrc = (const char*)(h_in + (size_t)row * Uu + cb);
            char*       ldst = (char*)(hA + row * HROW + cb);
#ifdef ASYNC_LDS
#pragma unroll
            for (int q = 0; q < 8; ++q)
                __builtin_amdgcn_global_load_async_to_lds_b128(
                    (v4i_vs*)(gsrc + q * 16),
                    (v4i_vs*)(ldst + q * 16),
                    0, 0);
#else
            const uint4* s = (const uint4*)gsrc;
            uint4*       d = (uint4*)ldst;
#pragma unroll
            for (int q = 0; q < 8; ++q) d[q] = s[q];
#endif
        }

        // ---- acc = Zpre + bias (overlaps the async staging) ; prefetch t+1 ----
        v8f acc;
#pragma unroll
        for (int v = 0; v < 8; ++v) {
            int brow = mt * 16 + hi * 8 + v;  // batch index
            const _Float16* zp = Z + ((size_t)brow * Tt + t) * G4 + ncol + r;
            acc[v] = (float)(*zp) + bval;
            if (t + 1 < Tt)
                __builtin_prefetch(zp + G4, 0, 3);   // -> global_prefetch_b8
        }

#ifdef ASYNC_LDS
        __builtin_amdgcn_s_wait_asynccnt(0);
#endif
        __syncthreads();

        // ---- z += h_{t-1} @ U : double-buffered LDS fragments ----
        {
            v16h a0, b0, a1, b1;
            load_frag(0, a0, b0);
#pragma unroll 2
            for (int kt = 0; kt < 14; kt += 2) {
                load_frag(kt + 1, a1, b1);
                acc = __builtin_amdgcn_wmma_f32_16x16x32_f16(false, a0, false, b0,
                                                             (short)0, acc, false, false);
                load_frag(kt + 2, a0, b0);
                acc = __builtin_amdgcn_wmma_f32_16x16x32_f16(false, a1, false, b1,
                                                             (short)0, acc, false, false);
            }
            load_frag(15, a1, b1);
            acc = __builtin_amdgcn_wmma_f32_16x16x32_f16(false, a0, false, b0,
                                                         (short)0, acc, false, false);
            acc = __builtin_amdgcn_wmma_f32_16x16x32_f16(false, a1, false, b1,
                                                         (short)0, acc, false, false);
        }

#pragma unroll
        for (int v = 0; v < 8; ++v)
            zbuf[wave * 256 + (hi * 8 + v) * 16 + r] = acc[v];
        __syncthreads();

        // ---- 512 cell updates (32 batch x 16 units), 2 per thread ----
#pragma unroll
        for (int k = 0; k < 2; ++k) {
            int idx = threadIdx.x + k * 256;
            int j = idx >> 5;                 // unit within block
            int bidx = idx & 31;              // batch
            int mtt = bidx >> 4, br = bidx & 15;
            float zi = zbuf[(0 * 2 + mtt) * 256 + br * 16 + j];
            float zf = zbuf[(1 * 2 + mtt) * 256 + br * 16 + j];
            float zg = zbuf[(2 * 2 + mtt) * 256 + br * 16 + j];
            float zo = zbuf[(3 * 2 + mtt) * 256 + br * 16 + j];
            int u = blockIdx.x * 16 + j;
            float c_old = c_state[(size_t)bidx * Uu + u];
            float ig = 1.0f / (1.0f + __expf(-zi));
            float fg = 1.0f / (1.0f + __expf(-zf));
            float gg = tanhf(zg);
            float og = 1.0f / (1.0f + __expf(-zo));
            float c_new = fg * c_old + ig * gg;
            float h_new = og * tanhf(c_new);
            c_state[(size_t)bidx * Uu + u] = c_new;
            _Float16 hh = (_Float16)h_new;
            h_out[(size_t)bidx * Uu + u] = hh;
            Hseq[((size_t)bidx * Tt + t) * Uu + u] = hh;
        }

        // ---- device-wide step barrier (monotonic counter, no reset race) ----
        __threadfence();                      // publish h_out before arriving
        __syncthreads();                      // whole block done with step t
        if (threadIdx.x == 0) {
            __hip_atomic_fetch_add(barrier_ctr, 1u, __ATOMIC_RELEASE,
                                   __HIP_MEMORY_SCOPE_AGENT);
            unsigned int target = (unsigned int)NBLK * (unsigned int)(t + 1);
            while (__hip_atomic_load(barrier_ctr, __ATOMIC_ACQUIRE,
                                     __HIP_MEMORY_SCOPE_AGENT) < target) {
                __builtin_amdgcn_s_sleep(1);
            }
        }
        __syncthreads();                      // release whole block into t+1
        __threadfence();
    }
}

// ---------------------------------------------------------------------------
// Host driver
// ---------------------------------------------------------------------------
extern "C" void kernel_launch(void* const* d_in, const int* in_sizes, int n_in,
                              void* d_out, int out_size, void* d_ws, size_t ws_size,
                              hipStream_t stream) {
    (void)in_sizes; (void)n_in; (void)out_size; (void)ws_size;

    const float* x   = (const float*)d_in[0];
    const float* W0  = (const float*)d_in[1];
    const float* U0  = (const float*)d_in[2];
    const float* b0  = (const float*)d_in[3];
    const float* h00 = (const float*)d_in[4];
    const float* c00 = (const float*)d_in[5];
    const float* W1  = (const float*)d_in[6];
    const float* U1  = (const float*)d_in[7];
    const float* b1  = (const float*)d_in[8];
    const float* h01 = (const float*)d_in[9];
    const float* c01 = (const float*)d_in[10];
    const float* W2  = (const float*)d_in[11];
    const float* U2  = (const float*)d_in[12];
    const float* b2  = (const float*)d_in[13];
    const float* h02 = (const float*)d_in[14];
    const float* c02 = (const float*)d_in[15];
    const float* Wd  = (const float*)d_in[16];
    const float* bd  = (const float*)d_in[17];

    char* base = (char*)d_ws;
    size_t off = 0;
    auto carve = [&](size_t bytes) -> void* {
        void* q = base + off;
        off = (off + bytes + 255) & ~(size_t)255;
        return q;
    };

    _Float16* Wp0 = (_Float16*)carve((size_t)Dd * G4 * 2);
    _Float16* Wp1 = (_Float16*)carve((size_t)Uu * G4 * 2);
    _Float16* Wp2 = (_Float16*)carve((size_t)Uu * G4 * 2);
    _Float16* Up0 = (_Float16*)carve((size_t)Uu * G4 * 2);
    _Float16* Up1 = (_Float16*)carve((size_t)Uu * G4 * 2);
    _Float16* Up2 = (_Float16*)carve((size_t)Uu * G4 * 2);
    _Float16* Wpd = (_Float16*)carve((size_t)Uu * Dd * 2);
    _Float16* Z   = (_Float16*)carve((size_t)MROWS * G4 * 2);
    _Float16* H0  = (_Float16*)carve((size_t)MROWS * Uu * 2);
    _Float16* H1  = (_Float16*)carve((size_t)MROWS * Uu * 2);
    _Float16* H2  = H0;  // H0 is dead once Z1 = H0@W1 is computed
    _Float16* hb0 = (_Float16*)carve((size_t)Bb * Uu * 2);
    _Float16* hb1 = (_Float16*)carve((size_t)Bb * Uu * 2);
    float*    cst = (float*)carve((size_t)Bb * Uu * 4);
    unsigned int* bar = (unsigned int*)carve(256);   // barrier counter, own line

    auto pack = [&](const float* W, _Float16* out, int K, int N) {
        int tiles = (K >> 5) * (N >> 4);
        pack_b<<<(tiles + 7) / 8, 256, 0, stream>>>(W, out, K, N);
    };
    pack(W0, Wp0, Dd, G4);
    pack(W1, Wp1, Uu, G4);
    pack(W2, Wp2, Uu, G4);
    pack(U0, Up0, Uu, G4);
    pack(U1, Up1, Uu, G4);
    pack(U2, Up2, Uu, G4);
    pack(Wd, Wpd, Uu, Dd);

    auto run_layer = [&](const _Float16* Zl, const _Float16* Up, const float* bias,
                         const float* h0v, const float* c0v, _Float16* Hseq) {
        init_state<<<(Bb * Uu) / 256, 256, 0, stream>>>(h0v, c0v, hb0, cst, bar);
        lstm_layer<<<NBLK, 256, 0, stream>>>(Zl, Up, bias, hb0, hb1, cst, Hseq, bar);
    };

    const int tilesZ = (MROWS >> 4) * (G4 >> 4);   // 131072
    const int tilesO = (MROWS >> 4) * (Dd >> 4);   // 16384

    // Layer 0
    gemm_wmma<true, true><<<tilesZ / 8, 256, 0, stream>>>(
        (const void*)x, Wp0, nullptr, (void*)Z, MROWS, G4, Dd, Dd);
    run_layer(Z, Up0, b0, h00, c00, H0);

    // Layer 1
    gemm_wmma<false, true><<<tilesZ / 8, 256, 0, stream>>>(
        (const void*)H0, Wp1, nullptr, (void*)Z, MROWS, G4, Uu, Uu);
    run_layer(Z, Up1, b1, h01, c01, H1);

    // Layer 2
    gemm_wmma<false, true><<<tilesZ / 8, 256, 0, stream>>>(
        (const void*)H1, Wp2, nullptr, (void*)Z, MROWS, G4, Uu, Uu);
    run_layer(Z, Up2, b2, h02, c02, H2);

    // Dense head: d_out = H2 @ Wd + bd  (f32 out)
    gemm_wmma<false, false><<<tilesO / 8, 256, 0, stream>>>(
        (const void*)H2, Wpd, bd, d_out, MROWS, Dd, Uu, Uu);
}
